// Attention_44006234915573
// MI455X (gfx1250) — matrix operand order repaired and checked by reading, not compiled, not run on hardware
//
#include <hip/hip_runtime.h>

#define NTOK 625
#define NPAD 640
#define DIM 128
#define DH 32
#define HEADS 4
#define BATCH 256
#define SCALE_F 0.17677669529663687f

typedef __attribute__((ext_vector_type(16))) __bf16 v16bf;
typedef __attribute__((ext_vector_type(8)))  float  v8f;
typedef __attribute__((ext_vector_type(4)))  unsigned int v4u;

union FragU { v16bf v; v4u q[2]; };

// A-matrix fragment (16x32, 16-bit): lane<16 -> K{0..7,16..23}, lane>=16 -> K{8..15,24..31}
__device__ inline v16bf load_fragA(const __bf16* p, int hi) {
  FragU f;
  f.q[0] = *(const v4u*)(p + hi * 8);
  f.q[1] = *(const v4u*)(p + 16 + hi * 8);
  return f.v;
}
// B-matrix fragment (32x16, 16-bit): lane<16 -> K{0..7,16..23}, lane>=16 -> K{8..15,24..31}
__device__ inline v16bf load_fragB(const __bf16* p, int hi) {
  FragU f;
  f.q[0] = *(const v4u*)(p + hi * 8);
  f.q[1] = *(const v4u*)(p + 16 + hi * 8);
  return f.v;
}

__device__ inline unsigned short bfbits(float x) {
  union { float f; unsigned u; } a; a.f = x;
  unsigned u = a.u + 0x7fffu + ((a.u >> 16) & 1u);   // round-to-nearest-even
  return (unsigned short)(u >> 16);
}
__device__ inline __bf16 f2bf(float x) {
  union { unsigned short s; __bf16 b; } r; r.s = bfbits(x);
  return r.b;
}

__device__ inline v8f wmma_bf16(v16bf a, v16bf b, v8f c) {
  return __builtin_amdgcn_wmma_f32_16x16x32_bf16(false, a, false, b, (short)0, c, false, false);
}

// ---------------- weight conversion ----------------
__global__ void cvt_kernel(const float* __restrict__ in, __bf16* __restrict__ out, int n) {
  int i = blockIdx.x * blockDim.x + threadIdx.x;
  if (i < n) out[i] = f2bf(in[i]);
}

// ---------------- bias materialization: biasM[h][i_pad][j_pad], L2-resident ----------------
__global__ void biasx_kernel(const float* __restrict__ bt, const int* __restrict__ rel,
                             float* __restrict__ biasM) {
  int j = blockIdx.x * 128 + threadIdx.x;   // 0..639
  int i = blockIdx.y;                       // 0..639
  int h = blockIdx.z;
  float v;
  if (j >= NTOK)      v = -1e30f;           // pad cols kill softmax weight
  else if (i >= NTOK) v = 0.0f;             // pad rows: anything finite
  else                v = bt[(size_t)rel[(size_t)i * NTOK + j] * HEADS + h];
  biasM[((size_t)h * NPAD + i) * NPAD + j] = v;
}

// ---------------- x (b,d,n) -> xT bf16 (b, n_pad, d), pad rows zeroed ----------------
__global__ void xpose_kernel(const float* __restrict__ x, __bf16* __restrict__ xT) {
  __shared__ float tile[32][33];
  int b = blockIdx.z;
  int n0 = blockIdx.x * 32;
  int d0 = blockIdx.y * 32;
  const float* src = x + (size_t)b * DIM * NTOK;
  __bf16* dst = xT + (size_t)b * NPAD * DIM;
  int tn = threadIdx.x, td = threadIdx.y;
  #pragma unroll
  for (int i = 0; i < 32; i += 8) {
    int d = d0 + td + i, n = n0 + tn;
    tile[td + i][tn] = (n < NTOK) ? src[(size_t)d * NTOK + n] : 0.0f;
  }
  __syncthreads();
  #pragma unroll
  for (int i = 0; i < 32; i += 8) {
    int n = n0 + td + i, d = d0 + tn;
    dst[(size_t)n * DIM + d] = f2bf(tile[tn][td + i]);
  }
}

// ---------------- qkv = xT @ Wqkv^T ; split-store Q (scaled), K, V^T ----------------
__global__ __launch_bounds__(32) void qkv_kernel(const __bf16* __restrict__ xT,
    const __bf16* __restrict__ Wq, __bf16* __restrict__ Q, __bf16* __restrict__ K,
    __bf16* __restrict__ VT) {
  __shared__ __bf16 s_tb[16][16];
  int lane = threadIdx.x, hi = lane >> 4, lm = lane & 15;
  int m0 = blockIdx.x * 16;
  int b  = blockIdx.y;
  const __bf16* arow = xT + (size_t)b * NPAD * DIM + (size_t)(m0 + lm) * DIM;
  v16bf af[4];
  #pragma unroll
  for (int kk = 0; kk < 4; ++kk) af[kk] = load_fragA(arow + kk * 32, hi);

  for (int ot = 0; ot < 24; ++ot) {
    int o0 = ot * 16;                       // wave-uniform
    const __bf16* brow = Wq + (size_t)(o0 + lm) * DIM;
    v16bf bfr[4];                           // all loads in flight before WMMAs
    #pragma unroll
    for (int kk = 0; kk < 4; ++kk) bfr[kk] = load_fragB(brow + kk * 32, hi);
    v8f c = {};
    #pragma unroll
    for (int kk = 0; kk < 4; ++kk) c = wmma_bf16(af[kk], bfr[kk], c);

    int part = o0 >> 7;                     // scalar: 0=q, 1=k, 2=v
    int h    = (o0 & 127) >> 5;             // scalar head
    int dh0  = o0 & 31;                     // scalar: 0 or 16
    size_t hb = (size_t)b * HEADS + h;
    if (part < 2) {
      // LDS transpose -> row-contiguous coalesced stores
      float sc = (part == 0) ? SCALE_F : 1.0f;
      __bf16* dst = ((part == 0) ? Q : K) + hb * NPAD * DH;
      #pragma unroll
      for (int r = 0; r < 8; ++r) s_tb[r + hi * 8][lm] = f2bf(c[r] * sc);
      __syncthreads();
      #pragma unroll
      for (int rr = 0; rr < 8; ++rr) {
        int n = m0 + hi * 8 + rr;
        dst[(size_t)n * DH + dh0 + lm] = s_tb[hi * 8 + rr][lm];
      }
      __syncthreads();
    } else {
      // V^T: lane's 8 results are token-consecutive -> one packed b128 store
      v4u pk;
      #pragma unroll
      for (int r = 0; r < 4; ++r)
        pk[r] = (unsigned)bfbits(c[2 * r]) | ((unsigned)bfbits(c[2 * r + 1]) << 16);
      *(v4u*)&VT[(hb * DH + dh0 + lm) * NPAD + m0 + hi * 8] = pk;
    }
  }
}

// ---------------- attention: scores + bias -> softmax -> P @ V ----------------
__global__ __launch_bounds__(32) void attn_kernel(const __bf16* __restrict__ Q,
    const __bf16* __restrict__ K, const __bf16* __restrict__ VT,
    const float* __restrict__ biasM, __bf16* __restrict__ O) {
  __shared__ float  s_sim[16 * NPAD];   // 40 KB
  __shared__ __bf16 s_p[16 * NPAD];     // 20 KB, unnormalized exp
  __shared__ float  s_inv[16];
  __shared__ __bf16 s_to[16][32];
  int lane = threadIdx.x, hi = lane >> 4, lm = lane & 15;
  int i0 = blockIdx.x * 16, h = blockIdx.y, b = blockIdx.z;
  size_t hb = (size_t)b * HEADS + h;
  const __bf16* Qh = Q + hb * NPAD * DH;
  const __bf16* Kh = K + hb * NPAD * DH;
  const __bf16* Vh = VT + hb * DH * NPAD;
  const float*  bM = biasM + (size_t)h * NPAD * NPAD;

  // scores: one WMMA per 16x16 tile; double-buffered K fragments; branch-free bias add
  v16bf aq = load_fragA(Qh + (size_t)(i0 + lm) * DH, hi);
  v16bf bk = load_fragB(Kh + (size_t)lm * DH, hi);
  for (int jt = 0; jt < NPAD / 16; ++jt) {
    int j0 = jt * 16;
    v16bf bcur = bk;
    if (jt + 1 < NPAD / 16) bk = load_fragB(Kh + (size_t)(j0 + 16 + lm) * DH, hi);
    v8f c = {};
    c = wmma_bf16(aq, bcur, c);
    int j = j0 + lm;
    #pragma unroll
    for (int r = 0; r < 8; ++r) {
      int il = r + hi * 8;
      int i  = i0 + il;
      s_sim[il * NPAD + j] = c[r] + bM[(size_t)i * NPAD + j];  // coalesced, no branches
    }
  }
  __syncthreads();

  // softmax stats: float4 reads, single exp pass, packed b64 P stores,
  // normalization deferred to the P@V epilogue (row scale)
  {
    int row = lm;
    int jb = hi * (NPAD / 2);
    const float4* s4 = (const float4*)&s_sim[row * NPAD + jb];
    float mx = -1e30f;
    for (int j = 0; j < NPAD / 8; ++j) {
      float4 t = s4[j];
      mx = fmaxf(mx, fmaxf(fmaxf(t.x, t.y), fmaxf(t.z, t.w)));
    }
    mx = fmaxf(mx, __shfl_xor(mx, 16, 32));
    float sum = 0.0f;
    unsigned long long* p8 = (unsigned long long*)&s_p[row * NPAD + jb];
    for (int j = 0; j < NPAD / 8; ++j) {
      float4 t = s4[j];
      float e0 = __expf(t.x - mx), e1 = __expf(t.y - mx);
      float e2 = __expf(t.z - mx), e3 = __expf(t.w - mx);
      sum += (e0 + e1) + (e2 + e3);
      p8[j] = (unsigned long long)bfbits(e0)
            | ((unsigned long long)bfbits(e1) << 16)
            | ((unsigned long long)bfbits(e2) << 32)
            | ((unsigned long long)bfbits(e3) << 48);
    }
    sum += __shfl_xor(sum, 16, 32);
    if (lane < 16) s_inv[row] = 1.0f / sum;
  }
  __syncthreads();

  // out = P @ V  (two 16-wide N tiles, K padded to 640 with zeros), double-buffered frags
  v8f c0 = {}, c1 = {};
  v16bf ap  = load_fragA(&s_p[lm * NPAD], hi);
  v16bf vb0 = load_fragB(Vh + (size_t)lm * NPAD, hi);
  v16bf vb1 = load_fragB(Vh + (size_t)(16 + lm) * NPAD, hi);
  for (int kt = 0; kt < NPAD / 32; ++kt) {
    v16bf a_c = ap, b0_c = vb0, b1_c = vb1;
    int jn = (kt + 1) * 32;
    if (kt + 1 < NPAD / 32) {
      ap  = load_fragA(&s_p[lm * NPAD + jn], hi);
      vb0 = load_fragB(Vh + (size_t)lm * NPAD + jn, hi);
      vb1 = load_fragB(Vh + (size_t)(16 + lm) * NPAD + jn, hi);
    }
    c0 = wmma_bf16(a_c, b0_c, c0);
    c1 = wmma_bf16(a_c, b1_c, c1);
  }
  #pragma unroll
  for (int r = 0; r < 8; ++r) {
    float sc = s_inv[r + hi * 8];
    s_to[r + hi * 8][lm]      = f2bf(c0[r] * sc);
    s_to[r + hi * 8][16 + lm] = f2bf(c1[r] * sc);
  }
  __syncthreads();
  __bf16* Ob = O + (size_t)b * NPAD * DIM;
  #pragma unroll
  for (int rr = 0; rr < 16; ++rr) {
    int n = i0 + rr;
    Ob[(size_t)n * DIM + h * DH + lane] = s_to[rr][lane];   // 64B coalesced per row
  }
}

// ---------------- out = O @ Wout^T, LDS-transposed coalesced store to (b,d,n) fp32 ----------------
__global__ __launch_bounds__(32) void proj_kernel(const __bf16* __restrict__ O,
    const __bf16* __restrict__ Wo, float* __restrict__ out) {
  __shared__ float s_t[16][17];
  int lane = threadIdx.x, hi = lane >> 4, lm = lane & 15;
  int m0 = blockIdx.x * 16, b = blockIdx.y;
  const __bf16* arow = O + (size_t)b * NPAD * DIM + (size_t)(m0 + lm) * DIM;
  float* outb = out + (size_t)b * DIM * NTOK;
  v16bf af[4];
  #pragma unroll
  for (int kk = 0; kk < 4; ++kk) af[kk] = load_fragA(arow + kk * 32, hi);

  for (int ot = 0; ot < DIM / 16; ++ot) {
    int o0 = ot * 16;
    const __bf16* brow = Wo + (size_t)(o0 + lm) * DIM;
    v16bf bfr[4];
    #pragma unroll
    for (int kk = 0; kk < 4; ++kk) bfr[kk] = load_fragB(brow + kk * 32, hi);
    v8f c = {};
    #pragma unroll
    for (int kk = 0; kk < 4; ++kk) c = wmma_bf16(af[kk], bfr[kk], c);
    #pragma unroll
    for (int r = 0; r < 8; ++r) s_t[r + hi * 8][lm] = c[r];   // s_t[token][d]
    __syncthreads();
    #pragma unroll
    for (int rr = 0; rr < 8; ++rr) {
      int d = o0 + hi * 8 + rr;
      int n = m0 + lm;
      if (n < NTOK) outb[(size_t)d * NTOK + n] = s_t[lm][hi * 8 + rr]; // 64B runs
    }
    __syncthreads();
  }
}

extern "C" void kernel_launch(void* const* d_in, const int* in_sizes, int n_in,
                              void* d_out, int out_size, void* d_ws, size_t ws_size,
                              hipStream_t stream) {
  (void)in_sizes; (void)n_in; (void)out_size; (void)ws_size;
  const float* x     = (const float*)d_in[0];
  const float* Wqkv  = (const float*)d_in[1];
  const float* Wout  = (const float*)d_in[2];
  const float* bias  = (const float*)d_in[3];
  const int*   rel   = (const int*)d_in[4];
  float* out = (float*)d_out;

  char* ws = (char*)d_ws;
  size_t off = 0;
  __bf16* wq_bf = (__bf16*)(ws + off); off += (size_t)3 * DIM * DIM * 2;
  __bf16* wo_bf = (__bf16*)(ws + off); off += (size_t)DIM * DIM * 2;
  float*  biasM = (float*)(ws + off);  off += (size_t)HEADS * NPAD * NPAD * 4;
  __bf16* xT    = (__bf16*)(ws + off); off += (size_t)BATCH * NPAD * DIM * 2;
  __bf16* Q     = (__bf16*)(ws + off); off += (size_t)BATCH * HEADS * NPAD * DH * 2;
  __bf16* K     = (__bf16*)(ws + off); off += (size_t)BATCH * HEADS * NPAD * DH * 2;
  __bf16* VT    = (__bf16*)(ws + off); off += (size_t)BATCH * HEADS * DH * NPAD * 2;
  __bf16* O     = xT;   // xT is dead after qkv_kernel -> reuse as attention output

  cvt_kernel<<<(3 * DIM * DIM + 255) / 256, 256, 0, stream>>>(Wqkv, wq_bf, 3 * DIM * DIM);
  cvt_kernel<<<(DIM * DIM + 255) / 256, 256, 0, stream>>>(Wout, wo_bf, DIM * DIM);
  biasx_kernel<<<dim3(NPAD / 128, NPAD, HEADS), 128, 0, stream>>>(bias, rel, biasM);
  xpose_kernel<<<dim3(NPAD / 32, DIM / 32, BATCH), dim3(32, 8), 0, stream>>>(x, xT);
  qkv_kernel<<<dim3(NPAD / 16, BATCH), 32, 0, stream>>>(xT, wq_bf, Q, K, VT);
  attn_kernel<<<dim3(NPAD / 16, HEADS, BATCH), 32, 0, stream>>>(Q, K, VT, biasM, O);
  proj_kernel<<<dim3(NPAD / 16, BATCH), 32, 0, stream>>>(O, wo_bf, out);
}
